// ChannelMoEBlock_824633721568
// MI455X (gfx1250) — matrix operand order repaired and checked by reading, not compiled, not run on hardware
//
#include <hip/hip_runtime.h>
#include <hip/hip_bf16.h>
#include <math.h>

typedef __attribute__((ext_vector_type(16))) _Float16 v16h;
typedef __attribute__((ext_vector_type(8)))  float    v8f;

#define B_ 8
#define S_ 4096
#define C_ 768
#define E_ 8
#define K_ 192
#define H_ 768
#define LN_EPS 1e-5f
#define MTILE 128
#define NTHREADS 256
#define HCHUNKS 24               // H / 32
#define FRAG_HALVES 512          // 32 lanes * 16 halves
#define CHUNK_HALVES (12*FRAG_HALVES)    // 6144 halves = 12 KB (one matrix)
#define PAIR_HALVES  (2*CHUNK_HALVES)    // w1 chunk + w2 chunk = 24 KB
#define LDSB 24576               // one ping-pong buffer (bytes)

// ---------------- kernel 1: split-S channel stats ----------------
__global__ void k_stats(const float* __restrict__ x, float* __restrict__ sums) {
    // grid (B*3, 16), block 256
    int b  = blockIdx.x / (C_/256);
    int ct = blockIdx.x % (C_/256);
    int c  = ct*256 + threadIdx.x;
    int s0 = blockIdx.y * (S_/16);
    const float* px = x + ((size_t)b*S_ + s0)*C_ + c;
    float s1 = 0.f, s2 = 0.f;
    for (int s = 0; s < S_/16; ++s) { float v = px[(size_t)s*C_]; s1 += v; s2 += v*v; }
    atomicAdd(&sums[(b*C_+c)*2+0], s1);
    atomicAdd(&sums[(b*C_+c)*2+1], s2);
}

// ---------------- kernel 2: gate features + softmax ----------------
__global__ void k_gate(const float* __restrict__ sums,
                       const float* __restrict__ ln_g, const float* __restrict__ ln_b,
                       const float* __restrict__ ep_w, const float* __restrict__ ep_b,
                       const float* __restrict__ gl_w, const float* __restrict__ gl_b,
                       float* __restrict__ aff) {
    int i = blockIdx.x*blockDim.x + threadIdx.x;   // over B*C
    if (i >= B_*C_) return;
    float mean   = sums[i*2+0] * (1.0f/S_);
    float energy = sqrtf(sums[i*2+1] * (1.0f/S_));
    // LayerNorm over a length-1 axis: (v - mean(v)) == 0
    float v  = mean, mu = v;
    float fd = ((v - mu) * rsqrtf(0.0f + LN_EPS)) * ln_g[0] + ln_b[0];
    float fe = energy * ep_w[0] + ep_b[0];
    float lg[E_]; float mx = -1e30f;
    #pragma unroll
    for (int e = 0; e < E_; ++e) {
        lg[e] = gl_w[e*2+0]*fd + gl_w[e*2+1]*fe + gl_b[e];
        mx = fmaxf(mx, lg[e]);
    }
    float den = 0.f;
    #pragma unroll
    for (int e = 0; e < E_; ++e) { lg[e] = expf(lg[e]-mx); den += lg[e]; }
    float inv = 1.0f/den;
    #pragma unroll
    for (int e = 0; e < E_; ++e) aff[(size_t)i*E_ + e] = lg[e]*inv;
}

// ---------------- kernel 3: exact stable top-K per (b,e) ----------------
__global__ __launch_bounds__(C_) void k_topk(const float* __restrict__ aff,
                                             int* __restrict__ idxg,
                                             float* __restrict__ gateg,
                                             float* __restrict__ tw) {
    __shared__ float sval[C_];
    int be = blockIdx.x; int e = be % E_; int b = be / E_;
    int c = threadIdx.x;
    float v = aff[((size_t)b*C_ + c)*E_ + e];
    sval[c] = v;
    __syncthreads();
    int rank = 0;
    for (int j = 0; j < C_; ++j) {
        float vj = sval[j];
        rank += (vj > v) || (vj == v && j < c);   // stable: ties by lower index
    }
    if (rank < K_) {
        idxg [(size_t)be*K_ + rank] = c;
        gateg[(size_t)be*K_ + rank] = v;
        atomicAdd(&tw[b*C_ + c], v);
    }
}

// ---------------- kernel 4: weight normalization ----------------
__global__ void k_wt(const int* __restrict__ idxg, const float* __restrict__ gateg,
                     const float* __restrict__ tw, float* __restrict__ wt) {
    int i = blockIdx.x*blockDim.x + threadIdx.x;
    if (i >= B_*E_*K_) return;
    int b = i / (E_*K_);
    float t = fmaxf(tw[b*C_ + idxg[i]], 1e-8f);
    wt[i] = gateg[i] / t;
}

// ---------------- kernel 5: repack W1/W2 -> interleaved f16 B-fragment-major ----------------
// per (e,hc) chunk: [w1 frags (12x32x16)][w2 frags (12x32x16)] contiguous = 24 KB.
// fragment elem q -> K = (q<8 ? q : 16+q-8) + 8*(lane>=16), N = lane&15.
__global__ void k_repack(const float* __restrict__ W1, const float* __restrict__ W2,
                         _Float16* __restrict__ wfr) {
    int i = blockIdx.x*blockDim.x + threadIdx.x;
    const int total = E_*HCHUNKS*12*32*16;       // per matrix
    if (i >= total) return;
    int q  = i & 15;
    int l  = (i >> 4) & 31;
    int f  = (i >> 9) % 12;
    int t  = (i >> 9) / 12;
    int hc = t % HCHUNKS;
    int e  = t / HCHUNKS;
    int hh = l >> 4, n = l & 15;
    int kb = ((q < 8) ? q : 16 + (q - 8)) + 8*hh; // K inside a 32-row block
    int chunk = e*HCHUNKS + hc;
    int inner = i % CHUNK_HALVES;                // f*512 + l*16 + q
    {   // stage-1 B: W1 chunk is (192 x 32): frag = kk*2+nsub
        int kk = f >> 1, nsub = f & 1;
        int krow = kk*32 + kb;                   // 0..191
        int col  = hc*32 + nsub*16 + n;          // 0..767
        wfr[(size_t)chunk*PAIR_HALVES + inner] =
            (_Float16)W1[((size_t)e*K_ + krow)*H_ + col];
    }
    {   // stage-2 B: W2 chunk is (32 x 192): frag = n0
        int n0   = f;
        int hrow = hc*32 + kb;                   // 0..767
        int colk = n0*16 + n;                    // 0..191
        wfr[(size_t)chunk*PAIR_HALVES + CHUNK_HALVES + inner] =
            (_Float16)W2[((size_t)e*H_ + hrow)*K_ + colk];
    }
}

// issue one 24 KB chunk copy global->LDS via async-DMA path (ASYNCcnt tracked)
__device__ __forceinline__ void async_copy_chunk(const _Float16* gsrc,
                                                 unsigned lds_base, int tid) {
    #pragma unroll
    for (int i = 0; i < 6; ++i) {                // 256 thr * 6 * 16B = 24576 B
        unsigned ldsa = lds_base + (unsigned)(tid + NTHREADS*i)*16u;
        const char* ga = (const char*)gsrc + (size_t)(tid + NTHREADS*i)*16;
        asm volatile("global_load_async_to_lds_b128 %0, %1, off"
                     :: "v"(ldsa), "v"(ga) : "memory");
    }
}

// ---------------- kernel 6: main per-(b,e) MLP with WMMA ----------------
__global__ __launch_bounds__(NTHREADS) void k_moe(const float* __restrict__ x,
                                                  const _Float16* __restrict__ wfr,
                                                  const float* __restrict__ b1g,
                                                  const float* __restrict__ b2g,
                                                  const int* __restrict__ idxg,
                                                  const float* __restrict__ wtg,
                                                  float* __restrict__ out) {
    // LDS overlay: [0,49152)      = 128x192 f16 X tile, later ping buffers 0 and 1
    //              [49152,58368)  = per-wave h transpose (8x16x36 halves)
    //              [58368,59136)  = idx, [59136,59904) = wt
    __shared__ __align__(16) char smem[2*LDSB + 9216 + 768 + 768];
    _Float16* lds_x  = (_Float16*)smem;                      // [128][192]
    _Float16* lds_h  = (_Float16*)(smem + 2*LDSB);           // [8][16][36]
    int*      lds_idx= (int*)   (smem + 2*LDSB + 9216);
    float*    lds_wt = (float*) (smem + 2*LDSB + 9216 + 768);
    const unsigned lds0 = (unsigned)(size_t)smem;            // LDS byte offset (addr[31:0])

    const int tid  = threadIdx.x;
    const int bid  = blockIdx.x;
    const int e    = bid & 7;
    const int st   = (bid >> 3) & 31;
    const int b    = bid >> 8;
    const int row0 = st * MTILE;
    const int wave = tid >> 5, lane = tid & 31;
    const int hh   = lane >> 4, lm = lane & 15;
    const int be   = b*E_ + e;

    const _Float16* gw = wfr + (size_t)(e*HCHUNKS) * PAIR_HALVES;

    for (int i = tid; i < K_; i += NTHREADS) {
        lds_idx[i] = idxg[(size_t)be*K_ + i];
        lds_wt[i]  = wtg [(size_t)be*K_ + i];
    }
    __syncthreads();
    // gather X tile (128 rows x 192 selected channels), f32 -> f16
    for (int i = tid; i < MTILE*K_; i += NTHREADS) {
        int r = i / K_, ks = i - r*K_;
        lds_x[r*K_ + ks] = (_Float16)x[((size_t)b*S_ + row0 + r)*C_ + lds_idx[ks]];
    }
    __syncthreads();

    // invariant A fragments (16-bit A 16x32 layout, ISA 7.12.2), kept in VGPRs;
    // after this the X region becomes the two W ping-pong buffers.
    v16h afrag[6];
    #pragma unroll
    for (int kk = 0; kk < 6; ++kk) {
        #pragma unroll
        for (int q = 0; q < 16; ++q) {
            int v = q >> 1, par = q & 1;
            int k = kk*32 + ((v < 4) ? 2*v : 16 + 2*(v-4)) + par + hh*8;
            afrag[kk][q] = lds_x[(wave*16 + lm)*K_ + k];
        }
    }

    v8f yacc[12];
    #pragma unroll
    for (int n0 = 0; n0 < 12; ++n0) yacc[n0] = {};

    __syncthreads();                       // all waves done reading X region
    async_copy_chunk(gw, lds0, tid);       // chunk 0 -> buffer 0

    for (int hc = 0; hc < HCHUNKS; ++hc) {
        // make chunk hc visible to the whole block; everyone is also done
        // computing with chunk hc-1 once they pass this barrier.
        asm volatile("s_wait_asynccnt 0" ::: "memory");
        __syncthreads();
        const _Float16* cur_w1 = (const _Float16*)(smem + (hc&1)*LDSB);
        const _Float16* cur_w2 = cur_w1 + CHUNK_HALVES;
        if (hc + 1 < HCHUNKS)                        // prefetch next chunk into
            async_copy_chunk(gw + (size_t)(hc+1)*PAIR_HALVES,
                             lds0 + (unsigned)((hc+1)&1)*LDSB, tid);

        // ---- stage 1: H32 = gelu(X(16x192) @ W1chunk(192x32) + b1) ----
        #pragma unroll
        for (int nsub = 0; nsub < 2; ++nsub) {
            v16h bf[6];
            #pragma unroll
            for (int kk = 0; kk < 6; ++kk)
                bf[kk] = *(const v16h*)(cur_w1 + ((kk*2 + nsub)*32 + lane)*16);
            v8f hacc = {};
            #pragma unroll
            for (int kk = 0; kk < 6; ++kk)
                hacc = __builtin_amdgcn_wmma_f32_16x16x32_f16(false, afrag[kk],
                                                              false, bf[kk],
                                                              (short)0, hacc, false, false);
            int coln  = hc*32 + nsub*16 + lm;
            float bias = b1g[e*H_ + coln];
            #pragma unroll
            for (int j = 0; j < 8; ++j) {           // D layout: row = j + 8*hh, col = lm
                float t = hacc[j] + bias;
                float g = 0.5f*t*(1.0f + erff(t*0.70710678118654752f));  // exact gelu
                lds_h[(wave*16 + j + 8*hh)*36 + nsub*16 + lm] = (_Float16)g;
            }
        }
        // DS ops are in-order per wave; wait then re-read in A-fragment layout
        asm volatile("s_wait_dscnt 0" ::: "memory");

        v16h hfrag;
        #pragma unroll
        for (int q = 0; q < 16; ++q) {
            int v = q >> 1, par = q & 1;
            int k = ((v < 4) ? 2*v : 16 + 2*(v-4)) + par + hh*8;
            hfrag[q] = lds_h[(wave*16 + lm)*36 + k];
        }

        // ---- stage 2: Y(16x192) += H32(16x32) @ W2chunk(32x192) ----
        v16h gb[4];
        #pragma unroll
        for (int t = 0; t < 4; ++t)
            gb[t] = *(const v16h*)(cur_w2 + (t*32 + lane)*16);
        #pragma unroll
        for (int grp = 0; grp < 3; ++grp) {
            v16h nb[4];
            if (grp < 2) {
                #pragma unroll
                for (int t = 0; t < 4; ++t)
                    nb[t] = *(const v16h*)(cur_w2 + (((grp+1)*4 + t)*32 + lane)*16);
            }
            #pragma unroll
            for (int t = 0; t < 4; ++t)
                yacc[grp*4+t] = __builtin_amdgcn_wmma_f32_16x16x32_f16(false, hfrag,
                                                                       false, gb[t],
                                                                       (short)0, yacc[grp*4+t],
                                                                       false, false);
            if (grp < 2) {
                #pragma unroll
                for (int t = 0; t < 4; ++t) gb[t] = nb[t];
            }
        }
    }

    // ---- epilogue: +b2, *wt_norm, scatter-add into out ----
    #pragma unroll
    for (int n0 = 0; n0 < 12; ++n0) {
        int ks  = n0*16 + lm;
        float w = lds_wt[ks];
        float bv = b2g[e*K_ + ks];
        int c = lds_idx[ks];
        float* po = out + ((size_t)b*S_ + row0 + wave*16)*C_ + c;
        #pragma unroll
        for (int j = 0; j < 8; ++j) {
            int m = j + 8*hh;
            atomicAdd(po + (size_t)m*C_, (yacc[n0][j] + bv)*w);
        }
    }
}

extern "C" void kernel_launch(void* const* d_in, const int* in_sizes, int n_in,
                              void* d_out, int out_size, void* d_ws, size_t ws_size,
                              hipStream_t stream) {
    const float* x    = (const float*)d_in[0];
    const float* ln_g = (const float*)d_in[1];
    const float* ln_b = (const float*)d_in[2];
    const float* ep_w = (const float*)d_in[3];
    const float* ep_b = (const float*)d_in[4];
    const float* gl_w = (const float*)d_in[5];
    const float* gl_b = (const float*)d_in[6];
    const float* W1   = (const float*)d_in[7];
    const float* b1   = (const float*)d_in[8];
    const float* W2   = (const float*)d_in[9];
    const float* b2   = (const float*)d_in[10];
    float* out = (float*)d_out;

    char* ws = (char*)d_ws;
    float*    sums = (float*)(ws + 0);          //  49152 B
    float*    aff  = (float*)(ws + 49152);      // 196608 B
    float*    tw   = (float*)(ws + 245760);     //  24576 B
    int*      idxg = (int*)  (ws + 270336);     //  49152 B
    float*    gate = (float*)(ws + 319488);     //  49152 B
    float*    wt   = (float*)(ws + 368640);     //  49152 B
    _Float16* wfr  = (_Float16*)(ws + 417792);  // 4718592 B -> total ~5.1 MB

    hipMemsetAsync(sums, 0, 49152, stream);
    hipMemsetAsync(tw,   0, 24576, stream);
    hipMemsetAsync(out,  0, (size_t)out_size*sizeof(float), stream);

    dim3 gs(B_*(C_/256), 16);
    k_stats<<<gs, 256, 0, stream>>>(x, sums);
    k_gate<<<(B_*C_ + 255)/256, 256, 0, stream>>>(sums, ln_g, ln_b, ep_w, ep_b, gl_w, gl_b, aff);
    k_topk<<<B_*E_, C_, 0, stream>>>(aff, idxg, gate, tw);
    k_wt<<<(B_*E_*K_ + 255)/256, 256, 0, stream>>>(idxg, gate, tw, wt);
    const int total = E_*HCHUNKS*12*32*16;
    k_repack<<<(total + 255)/256, 256, 0, stream>>>(W1, W2, wfr);
    k_moe<<<B_*E_*(S_/MTILE), NTHREADS, 0, stream>>>(x, wfr, b1, b2, idxg, wt, out);
}
